// SuperGlue_78975858639532
// MI455X (gfx1250) — matrix operand order repaired and checked by reading, not compiled, not run on hardware
//
#include <hip/hip_runtime.h>

// ---------------------------------------------------------------------------
// CDNA5 (gfx1250) SuperGlue forward. wave32, WMMA f32<=f16 16x16x32 core.
// LDS tiles laid out so WMMA fragments are contiguous per lane:
//   As row-major   [m][k], stride 40 halfs -> A frag = 2x ds_load_b128
//   Bs transposed  [n][k], stride 40 halfs -> B frag = 2x ds_load_b128
// Global->LDS staging is register double-buffered: next K-tile is prefetched
// into VGPRs while the current tile's WMMAs execute.
// ---------------------------------------------------------------------------

typedef __attribute__((ext_vector_type(16))) _Float16 v16h;
typedef __attribute__((ext_vector_type(8)))  _Float16 v8h;
typedef __attribute__((ext_vector_type(4)))  _Float16 v4h;
typedef __attribute__((ext_vector_type(8)))  float    v8f;

#define BM 64
#define BN 128
#define BK 32
#define BKP 40   // padded K stride (halfs): 80B rows keep 16B alignment, spread banks
#define TPB 256

// ---------------------------------------------------------------------------
// WMMA inner tile: 256-thread block = 8 waves arranged 4(M) x 2(N).
// Each wave owns a 16x64 output strip -> 4 accumulators, 4 WMMAs per K-step.
// Fragment layouts per CDNA5 ISA 7.12.2:
//   A (16x32 f16): lane m=lane&15; elem e -> K=(e&7)+8*(lane>=16)+16*(e>=8)
//   B (32x16 f16): lane n=lane&15; elem e -> K=(e&7)+8*(lane>=16)+16*(e>=8)
//   C/D (16x16 f32): vgpr r -> M=r+8*(lane>=16), N=lane&15
// ---------------------------------------------------------------------------
__device__ __forceinline__ void wmma_tile(const _Float16* As, const _Float16* Bs,
                                          int waveM, int waveN, int lane, v8f* acc) {
  const int m  = lane & 15;
  const int kA = (lane >> 4) << 3;                 // 0 or 8
  const _Float16* ap = As + (waveM + m) * BKP + kA;
  v8h a0 = *(const v8h*)ap;                        // K = kA .. kA+7
  v8h a1 = *(const v8h*)(ap + 16);                 // K = kA+16 .. kA+23
  v16h a = __builtin_shufflevector(a0, a1, 0, 1, 2, 3, 4, 5, 6, 7,
                                            8, 9, 10, 11, 12, 13, 14, 15);
  const int n0 = lane & 15;
  const int kB = (lane >> 4) << 3;                 // 0 or 8
#pragma unroll
  for (int nt = 0; nt < 4; ++nt) {
    const _Float16* bp = Bs + (waveN + nt * 16 + n0) * BKP + kB;
    v8h b0 = *(const v8h*)bp;                      // K = kB .. kB+7
    v8h b1 = *(const v8h*)(bp + 16);               // K = kB+16 .. kB+23
    v16h b = __builtin_shufflevector(b0, b1, 0, 1, 2, 3, 4, 5, 6, 7,
                                              8, 9, 10, 11, 12, 13, 14, 15);
    acc[nt] = __builtin_amdgcn_wmma_f32_16x16x32_f16(
        false, a, false, b, (short)0, acc[nt], false, false);
  }
}

// ---------------------------------------------------------------------------
// Staging helpers (register prefetch <-> LDS store), one per access pattern.
// ---------------------------------------------------------------------------

// A row-major (weights / V / H): A[m,k] = A[m*lda + k]; float4 along K.
__device__ __forceinline__ void ldA_nn(const float* __restrict__ A, int lda,
                                       int blockM, int Mdim, int Kdim, int k0,
                                       int tid, float4* ra) {
#pragma unroll
  for (int i = 0; i < 2; ++i) {
    int g = tid + i * TPB;                         // 512 float4 groups
    int r = g >> 3, k4 = (g & 7) * 4;
    int gr = blockM + r, gk = k0 + k4;
    float4 v = {0.f, 0.f, 0.f, 0.f};
    if (gr < Mdim) {
      if (gk + 3 < Kdim) {
        v = *(const float4*)(A + (size_t)gr * lda + gk);
      } else {
        float* vp = (float*)&v;
#pragma unroll
        for (int j = 0; j < 4; ++j)
          if (gk + j < Kdim) vp[j] = A[(size_t)gr * lda + gk + j];
      }
    }
    ra[i] = v;
  }
}
__device__ __forceinline__ void stA_nn(_Float16* As, int tid, const float4* ra) {
#pragma unroll
  for (int i = 0; i < 2; ++i) {
    int g = tid + i * TPB;
    int r = g >> 3, k4 = (g & 7) * 4;
    v4h h; h[0] = (_Float16)ra[i].x; h[1] = (_Float16)ra[i].y;
           h[2] = (_Float16)ra[i].z; h[3] = (_Float16)ra[i].w;
    *(v4h*)(As + r * BKP + k4) = h;
  }
}

// A transposed (Q^T / md0^T): A[m,k] = A[k*sA + m]; float4 along M.
__device__ __forceinline__ void ldA_tn(const float* __restrict__ A, int sA,
                                       int blockM, int Mdim, int Kdim, int k0,
                                       int tid, float4* ra) {
#pragma unroll
  for (int i = 0; i < 2; ++i) {
    int g = tid + i * TPB;
    int k = g >> 4, r4 = (g & 15) * 4;
    int gm = blockM + r4, gk = k0 + k;
    float4 v = {0.f, 0.f, 0.f, 0.f};
    if (gk < Kdim && gm + 3 < Mdim)
      v = *(const float4*)(A + (size_t)gk * sA + gm);
    ra[i] = v;
  }
}
__device__ __forceinline__ void stA_tn(_Float16* As, int tid, const float4* ra) {
#pragma unroll
  for (int i = 0; i < 2; ++i) {
    int g = tid + i * TPB;
    int k = g >> 4, r4 = (g & 15) * 4;
    As[(r4 + 0) * BKP + k] = (_Float16)ra[i].x;
    As[(r4 + 1) * BKP + k] = (_Float16)ra[i].y;
    As[(r4 + 2) * BKP + k] = (_Float16)ra[i].z;
    As[(r4 + 3) * BKP + k] = (_Float16)ra[i].w;
  }
}

// B row-major K x N (activations): B[k,n]; float4 along N; split at ksplit.
__device__ __forceinline__ void ldB_kn(const float* __restrict__ B0,
                                       const float* __restrict__ B1, int ksplit,
                                       int ldb, int blockN, int Ndim, int Kdim,
                                       int k0, int tid, float4* rb) {
#pragma unroll
  for (int i = 0; i < 4; ++i) {
    int g = tid + i * TPB;                         // 1024 float4 groups
    int r = g >> 5, c4 = (g & 31) * 4;
    int gk = k0 + r, gc = blockN + c4;
    float4 v = {0.f, 0.f, 0.f, 0.f};
    if (gk < Kdim && gc + 3 < Ndim) {
      const float* Bp = (gk < ksplit) ? (B0 + (size_t)gk * ldb)
                                      : (B1 + (size_t)(gk - ksplit) * ldb);
      v = *(const float4*)(Bp + gc);
    }
    rb[i] = v;
  }
}
__device__ __forceinline__ void stB_kn(_Float16* Bs, int tid, const float4* rb) {
#pragma unroll
  for (int i = 0; i < 4; ++i) {
    int g = tid + i * TPB;
    int r = g >> 5, c4 = (g & 31) * 4;
    Bs[(c4 + 0) * BKP + r] = (_Float16)rb[i].x;
    Bs[(c4 + 1) * BKP + r] = (_Float16)rb[i].y;
    Bs[(c4 + 2) * BKP + r] = (_Float16)rb[i].z;
    Bs[(c4 + 3) * BKP + r] = (_Float16)rb[i].w;
  }
}

// B N x K rows (prob matrix accessed transposed): B[k,n] = B[n*sB + k].
__device__ __forceinline__ void ldB_nk(const float* __restrict__ B, int sB,
                                       int blockN, int Ndim, int Kdim, int k0,
                                       int tid, float4* rb) {
#pragma unroll
  for (int i = 0; i < 4; ++i) {
    int g = tid + i * TPB;
    int c = g >> 3, k4 = (g & 7) * 4;
    int gk = k0 + k4, gc = blockN + c;
    float4 v = {0.f, 0.f, 0.f, 0.f};
    if (gc < Ndim && gk + 3 < Kdim)
      v = *(const float4*)(B + (size_t)gc * sB + gk);
    rb[i] = v;
  }
}
__device__ __forceinline__ void stB_nk(_Float16* Bs, int tid, const float4* rb) {
#pragma unroll
  for (int i = 0; i < 4; ++i) {
    int g = tid + i * TPB;
    int c = g >> 3, k4 = (g & 7) * 4;
    v4h h; h[0] = (_Float16)rb[i].x; h[1] = (_Float16)rb[i].y;
           h[2] = (_Float16)rb[i].z; h[3] = (_Float16)rb[i].w;
    *(v4h*)(Bs + c * BKP + k4) = h;
  }
}

// ---------------------------------------------------------------------------
// GEMM (NN, weights): C = epi(A[M,K] @ B[K,N] + bias)
// ---------------------------------------------------------------------------
template <int EPI>
__global__ void __launch_bounds__(TPB) gemm_ws(
    const float* __restrict__ A, int lda,
    const float* __restrict__ B0, const float* __restrict__ B1, int ksplit, int ldb,
    const float* __restrict__ bias, const float* __restrict__ bns,
    const float* __restrict__ bnt, const float* __restrict__ resid,
    float* __restrict__ C, int ldc, int Mdim, int Ndim, int Kdim) {
  __shared__ __align__(16) _Float16 As[BM * BKP];
  __shared__ __align__(16) _Float16 Bs[BN * BKP];
  const int tid = threadIdx.x, lane = tid & 31, wid = tid >> 5;
  const int waveM = (wid >> 1) << 4, waveN = (wid & 1) << 6;
  const int blockM = blockIdx.x * BM, blockN = blockIdx.y * BN;
  v8f acc[4] = {};
  float4 ra[2], rb[4];
  ldA_nn(A, lda, blockM, Mdim, Kdim, 0, tid, ra);
  ldB_kn(B0, B1, ksplit, ldb, blockN, Ndim, Kdim, 0, tid, rb);
  for (int k0 = 0; k0 < Kdim; k0 += BK) {
    stA_nn(As, tid, ra);
    stB_kn(Bs, tid, rb);
    __syncthreads();
    if (k0 + BK < Kdim) {                          // prefetch next K-tile
      ldA_nn(A, lda, blockM, Mdim, Kdim, k0 + BK, tid, ra);
      ldB_kn(B0, B1, ksplit, ldb, blockN, Ndim, Kdim, k0 + BK, tid, rb);
    }
    wmma_tile(As, Bs, waveM, waveN, lane, acc);
    __syncthreads();
  }
#pragma unroll
  for (int nt = 0; nt < 4; ++nt) {
    int col = blockN + waveN + nt * 16 + (lane & 15);
#pragma unroll
    for (int r = 0; r < 8; ++r) {
      int row = blockM + waveM + r + ((lane >> 4) << 3);
      if (row < Mdim && col < Ndim) {
        float x = acc[nt][r] + bias[row];
        if (EPI == 1) { x = x * bns[row] + bnt[row]; x = fmaxf(x, 0.0f); }
        if (EPI == 2) { x += resid[(size_t)row * ldc + col]; }
        C[(size_t)row * ldc + col] = x;
      }
    }
  }
}

// ---------------------------------------------------------------------------
// GEMM (TN): C = scale * A^T B.  A[m,k] = A[k*sA + m], B[k,n] = B[k*sB + n].
// blockIdx.z batches over heads via zs* strides. Used for Q^T K and md0^T md1.
// ---------------------------------------------------------------------------
__global__ void __launch_bounds__(TPB) gemm_tn(
    const float* __restrict__ A, int sA, long zsA,
    const float* __restrict__ B, int sB, long zsB,
    float* __restrict__ C, int ldc, long zsC,
    int Mdim, int Ndim, int Kdim, float scale) {
  A += (size_t)blockIdx.z * zsA;
  B += (size_t)blockIdx.z * zsB;
  C += (size_t)blockIdx.z * zsC;
  __shared__ __align__(16) _Float16 As[BM * BKP];
  __shared__ __align__(16) _Float16 Bs[BN * BKP];
  const int tid = threadIdx.x, lane = tid & 31, wid = tid >> 5;
  const int waveM = (wid >> 1) << 4, waveN = (wid & 1) << 6;
  const int blockM = blockIdx.x * BM, blockN = blockIdx.y * BN;
  v8f acc[4] = {};
  float4 ra[2], rb[4];
  ldA_tn(A, sA, blockM, Mdim, Kdim, 0, tid, ra);
  ldB_kn(B, B, Kdim, sB, blockN, Ndim, Kdim, 0, tid, rb);
  for (int k0 = 0; k0 < Kdim; k0 += BK) {
    stA_tn(As, tid, ra);
    stB_kn(Bs, tid, rb);
    __syncthreads();
    if (k0 + BK < Kdim) {
      ldA_tn(A, sA, blockM, Mdim, Kdim, k0 + BK, tid, ra);
      ldB_kn(B, B, Kdim, sB, blockN, Ndim, Kdim, k0 + BK, tid, rb);
    }
    wmma_tile(As, Bs, waveM, waveN, lane, acc);
    __syncthreads();
  }
#pragma unroll
  for (int nt = 0; nt < 4; ++nt) {
    int col = blockN + waveN + nt * 16 + (lane & 15);
#pragma unroll
    for (int r = 0; r < 8; ++r) {
      int row = blockM + waveM + r + ((lane >> 4) << 3);
      if (row < Mdim && col < Ndim)
        C[(size_t)row * ldc + col] = acc[nt][r] * scale;
    }
  }
}

// ---------------------------------------------------------------------------
// GEMM (NT): C = A @ B^T-access.  A[m,k] = A[m*lda + k], B[k,n] = B[n*sB + k].
// Used for msg = V_h @ prob^T per head (blockIdx.z).
// ---------------------------------------------------------------------------
__global__ void __launch_bounds__(TPB) gemm_nt(
    const float* __restrict__ A, int lda, long zsA,
    const float* __restrict__ B, int sB, long zsB,
    float* __restrict__ C, int ldc, long zsC,
    int Mdim, int Ndim, int Kdim) {
  A += (size_t)blockIdx.z * zsA;
  B += (size_t)blockIdx.z * zsB;
  C += (size_t)blockIdx.z * zsC;
  __shared__ __align__(16) _Float16 As[BM * BKP];
  __shared__ __align__(16) _Float16 Bs[BN * BKP];
  const int tid = threadIdx.x, lane = tid & 31, wid = tid >> 5;
  const int waveM = (wid >> 1) << 4, waveN = (wid & 1) << 6;
  const int blockM = blockIdx.x * BM, blockN = blockIdx.y * BN;
  v8f acc[4] = {};
  float4 ra[2], rb[4];
  ldA_nn(A, lda, blockM, Mdim, Kdim, 0, tid, ra);
  ldB_nk(B, sB, blockN, Ndim, Kdim, 0, tid, rb);
  for (int k0 = 0; k0 < Kdim; k0 += BK) {
    stA_nn(As, tid, ra);
    stB_nk(Bs, tid, rb);
    __syncthreads();
    if (k0 + BK < Kdim) {
      ldA_nn(A, lda, blockM, Mdim, Kdim, k0 + BK, tid, ra);
      ldB_nk(B, sB, blockN, Ndim, Kdim, k0 + BK, tid, rb);
    }
    wmma_tile(As, Bs, waveM, waveN, lane, acc);
    __syncthreads();
  }
#pragma unroll
  for (int nt = 0; nt < 4; ++nt) {
    int col = blockN + waveN + nt * 16 + (lane & 15);
#pragma unroll
    for (int r = 0; r < 8; ++r) {
      int row = blockM + waveM + r + ((lane >> 4) << 3);
      if (row < Mdim && col < Ndim)
        C[(size_t)row * ldc + col] = acc[nt][r];
    }
  }
}

// ---------------------------------------------------------------------------
// Small kernels
// ---------------------------------------------------------------------------
__global__ void build_kin(const float* __restrict__ kp, const float* __restrict__ sc,
                          float* __restrict__ out) {
  int n = blockIdx.x * blockDim.x + threadIdx.x;
  if (n >= 1024) return;
  out[n]        = (kp[2 * n]     - 320.0f) * (1.0f / 448.0f);
  out[1024 + n] = (kp[2 * n + 1] - 240.0f) * (1.0f / 448.0f);
  out[2048 + n] = sc[n];
}

__global__ void softmax_rows(float* __restrict__ S, int cols) {
  float* p = S + (size_t)blockIdx.x * cols;
  __shared__ float red[TPB];
  int tid = threadIdx.x;
  float mx = -3.4e38f;
  for (int j = tid; j < cols; j += TPB) mx = fmaxf(mx, p[j]);
  red[tid] = mx; __syncthreads();
  for (int s = TPB / 2; s > 0; s >>= 1) {
    if (tid < s) red[tid] = fmaxf(red[tid], red[tid + s]);
    __syncthreads();
  }
  mx = red[0]; __syncthreads();
  float sum = 0.0f;
  for (int j = tid; j < cols; j += TPB) {
    float e = __expf(p[j] - mx); p[j] = e; sum += e;
  }
  red[tid] = sum; __syncthreads();
  for (int s = TPB / 2; s > 0; s >>= 1) {
    if (tid < s) red[tid] += red[tid + s];
    __syncthreads();
  }
  float inv = 1.0f / red[0];
  for (int j = tid; j < cols; j += TPB) p[j] *= inv;
}

__global__ void fill_bins(float* __restrict__ cpl, const float* __restrict__ bin) {
  float a = bin[0];
  int t = blockIdx.x * blockDim.x + threadIdx.x;
  if (t < 1024) {
    cpl[(size_t)t * 1025 + 1024] = a;
    cpl[(size_t)1024 * 1025 + t] = a;
  }
  if (t == 0) cpl[(size_t)1025 * 1025 - 1] = a;
}

__global__ void zero_uv(float* __restrict__ u, float* __restrict__ v) {
  int t = blockIdx.x * blockDim.x + threadIdx.x;
  if (t < 1025) { u[t] = 0.0f; v[t] = 0.0f; }
}

__global__ void sink_row(const float* __restrict__ cpl, const float* __restrict__ v,
                         float* __restrict__ u) {
  const int i = blockIdx.x;
  const float* row = cpl + (size_t)i * 1025;
  __shared__ float red[TPB];
  int tid = threadIdx.x;
  float mx = -3.4e38f;
  for (int j = tid; j < 1025; j += TPB) mx = fmaxf(mx, row[j] + v[j]);
  red[tid] = mx; __syncthreads();
  for (int s = TPB / 2; s > 0; s >>= 1) {
    if (tid < s) red[tid] = fmaxf(red[tid], red[tid + s]);
    __syncthreads();
  }
  mx = red[0]; __syncthreads();
  float sum = 0.0f;
  for (int j = tid; j < 1025; j += TPB) sum += __expf(row[j] + v[j] - mx);
  red[tid] = sum; __syncthreads();
  for (int s = TPB / 2; s > 0; s >>= 1) {
    if (tid < s) red[tid] += red[tid + s];
    __syncthreads();
  }
  if (tid == 0) {
    const float norm = -7.62461899f;               // -log(2048)
    float lmu = (i < 1024) ? norm : (6.93147181f + norm);
    u[i] = lmu - (mx + __logf(red[0]));
  }
}

__global__ void sink_col(const float* __restrict__ cpl, const float* __restrict__ u,
                         float* __restrict__ v) {
  int col = blockIdx.x * blockDim.x + threadIdx.x;
  if (col >= 1025) return;
  float mx = -3.4e38f, s = 0.0f;
  for (int i = 0; i < 1025; ++i) {
    float x = cpl[(size_t)i * 1025 + col] + u[i];
    if (x > mx) { s = s * __expf(mx - x) + 1.0f; mx = x; }
    else        { s += __expf(x - mx); }
  }
  const float norm = -7.62461899f;
  float lnu = (col < 1024) ? norm : (6.93147181f + norm);
  v[col] = lnu - (mx + __logf(s));
}

__global__ void row_argmax(const float* __restrict__ cpl, const float* __restrict__ u,
                           const float* __restrict__ v, float* __restrict__ mx0,
                           int* __restrict__ id0) {
  const int n = blockIdx.x;
  const float* row = cpl + (size_t)n * 1025;
  const float un = u[n] + 7.62461899f;             // Z = cpl + u + v - norm
  __shared__ float rv[TPB];
  __shared__ int   ri[TPB];
  int tid = threadIdx.x;
  float best = -3.4e38f; int bi = 0;
  for (int m = tid; m < 1024; m += TPB) {
    float z = row[m] + un + v[m];
    if (z > best) { best = z; bi = m; }
  }
  rv[tid] = best; ri[tid] = bi; __syncthreads();
  for (int s = TPB / 2; s > 0; s >>= 1) {
    if (tid < s) {
      if (rv[tid + s] > rv[tid] ||
          (rv[tid + s] == rv[tid] && ri[tid + s] < ri[tid])) {
        rv[tid] = rv[tid + s]; ri[tid] = ri[tid + s];
      }
    }
    __syncthreads();
  }
  if (tid == 0) { mx0[n] = rv[0]; id0[n] = ri[0]; }
}

__global__ void col_argmax(const float* __restrict__ cpl, const float* __restrict__ u,
                           const float* __restrict__ v, float* __restrict__ mx1,
                           int* __restrict__ id1) {
  const int m = blockIdx.x;
  const float vm = v[m] + 7.62461899f;
  __shared__ float rv[TPB];
  __shared__ int   ri[TPB];
  int tid = threadIdx.x;
  float best = -3.4e38f; int bi = 0;
  for (int i = tid; i < 1024; i += TPB) {
    float z = cpl[(size_t)i * 1025 + m] + u[i] + vm;
    if (z > best) { best = z; bi = i; }
  }
  rv[tid] = best; ri[tid] = bi; __syncthreads();
  for (int s = TPB / 2; s > 0; s >>= 1) {
    if (tid < s) {
      if (rv[tid + s] > rv[tid] ||
          (rv[tid + s] == rv[tid] && ri[tid + s] < ri[tid])) {
        rv[tid] = rv[tid + s]; ri[tid] = ri[tid + s];
      }
    }
    __syncthreads();
  }
  if (tid == 0) { mx1[m] = rv[0]; id1[m] = ri[0]; }
}

__global__ void match0(const float* __restrict__ mx0, const int* __restrict__ id0,
                       const int* __restrict__ id1, int* __restrict__ out0,
                       float* __restrict__ msc0) {
  int n = blockIdx.x * blockDim.x + threadIdx.x;
  if (n >= 1024) return;
  int j = id0[n];
  bool mutual = (id1[j] == n);
  float sc = mutual ? __expf(mx0[n]) : 0.0f;
  msc0[n] = sc;
  out0[n] = (mutual && sc > 0.2f) ? j : -1;
}

__global__ void match1(const int* __restrict__ id0, const int* __restrict__ id1,
                       const int* __restrict__ out0, const float* __restrict__ msc0,
                       int* __restrict__ out1, float* __restrict__ msc1) {
  int m = blockIdx.x * blockDim.x + threadIdx.x;
  if (m >= 1024) return;
  int i = id1[m];
  bool mutual = (id0[i] == m);
  msc1[m] = mutual ? msc0[i] : 0.0f;
  out1[m] = (mutual && out0[i] >= 0) ? i : -1;
}

// ---------------------------------------------------------------------------
// Host orchestration
// ---------------------------------------------------------------------------
extern "C" void kernel_launch(void* const* d_in, const int* in_sizes, int n_in,
                              void* d_out, int out_size, void* d_ws, size_t ws_size,
                              hipStream_t stream) {
  (void)in_sizes; (void)n_in; (void)out_size; (void)ws_size;
  const float* desc0 = (const float*)d_in[0];
  const float* desc1 = (const float*)d_in[1];
  const float* kp0   = (const float*)d_in[2];
  const float* kp1   = (const float*)d_in[3];
  const float* sc0   = (const float*)d_in[4];
  const float* sc1   = (const float*)d_in[5];
  const float* kw[4] = {(const float*)d_in[6],  (const float*)d_in[8],
                        (const float*)d_in[10], (const float*)d_in[12]};
  const float* kb[4] = {(const float*)d_in[7],  (const float*)d_in[9],
                        (const float*)d_in[11], (const float*)d_in[13]};
  const float* ks[3] = {(const float*)d_in[14], (const float*)d_in[16],
                        (const float*)d_in[18]};
  const float* kt[3] = {(const float*)d_in[15], (const float*)d_in[17],
                        (const float*)d_in[19]};
  const float* qw  = (const float*)d_in[20];
  const float* kwg = (const float*)d_in[21];
  const float* vw  = (const float*)d_in[22];
  const float* mw  = (const float*)d_in[23];
  const float* qb  = (const float*)d_in[24];
  const float* kbg = (const float*)d_in[25];
  const float* vbg = (const float*)d_in[26];
  const float* mbg = (const float*)d_in[27];
  const float* w1  = (const float*)d_in[28];
  const float* b1  = (const float*)d_in[29];
  const float* bns = (const float*)d_in[30];
  const float* bnt = (const float*)d_in[31];
  const float* w2  = (const float*)d_in[32];
  const float* b2  = (const float*)d_in[33];
  const float* fw  = (const float*)d_in[34];
  const float* fb  = (const float*)d_in[35];
  const float* bin = (const float*)d_in[36];

  float* ws = (float*)d_ws;
  size_t off = 0;
  auto alloc = [&](size_t n) { float* p = ws + off; off += (n + 63) & ~(size_t)63; return p; };
  float* xa[2]  = {alloc(262144), alloc(262144)};
  float* xb[2]  = {alloc(262144), alloc(262144)};
  float* kin    = alloc(3 * 1024);
  float* t32    = alloc(32 * 1024);
  float* t64    = alloc(64 * 1024);
  float* t128   = alloc(128 * 1024);
  float* Q      = alloc(262144);
  float* Kb     = alloc(262144);
  float* Vb     = alloc(262144);
  float* MSG    = alloc(262144);
  float* M2     = alloc(262144);
  float* Hb     = alloc(524288);
  float* P      = alloc(4ull * 1024 * 1024);
  float* md0    = alloc(262144);
  float* md1    = alloc(262144);
  float* cpl    = alloc(1025ull * 1025);
  float* uu     = alloc(1025);
  float* vv     = alloc(1025);
  float* mxr    = alloc(1024);
  float* mxc    = alloc(1024);
  int*   id0    = (int*)alloc(1024);
  int*   id1    = (int*)alloc(1024);

  // ---- keypoint encoder (3 -> 32 -> 64 -> 128 -> 256, + descriptors) ----
  auto kenc = [&](const float* kp, const float* sc, const float* desc, float* xout) {
    build_kin<<<4, TPB, 0, stream>>>(kp, sc, kin);
    gemm_ws<1><<<dim3(1, 8), TPB, 0, stream>>>(kw[0], 3, kin, kin, 3, 1024,
        kb[0], ks[0], kt[0], nullptr, t32, 1024, 32, 1024, 3);
    gemm_ws<1><<<dim3(1, 8), TPB, 0, stream>>>(kw[1], 32, t32, t32, 32, 1024,
        kb[1], ks[1], kt[1], nullptr, t64, 1024, 64, 1024, 32);
    gemm_ws<1><<<dim3(2, 8), TPB, 0, stream>>>(kw[2], 64, t64, t64, 64, 1024,
        kb[2], ks[2], kt[2], nullptr, t128, 1024, 128, 1024, 64);
    gemm_ws<2><<<dim3(4, 8), TPB, 0, stream>>>(kw[3], 128, t128, t128, 128, 1024,
        kb[3], nullptr, nullptr, desc, xout, 1024, 256, 1024, 128);
  };
  kenc(kp0, sc0, desc0, xa[0]);
  kenc(kp1, sc1, desc1, xb[0]);

  // ---- GNN: 18 alternating self/cross attention layers ----
  float* cur0 = xa[0]; float* nxt0 = xa[1];
  float* cur1 = xb[0]; float* nxt1 = xb[1];
  for (int l = 0; l < 18; ++l) {
    const float* qwl  = qw  + (size_t)l * 65536;
    const float* kwl  = kwg + (size_t)l * 65536;
    const float* vwl  = vw  + (size_t)l * 65536;
    const float* mwl  = mw  + (size_t)l * 65536;
    const float* qbl  = qb  + (size_t)l * 256;
    const float* kbl  = kbg + (size_t)l * 256;
    const float* vbl  = vbg + (size_t)l * 256;
    const float* mbl  = mbg + (size_t)l * 256;
    const float* w1l  = w1  + (size_t)l * 262144;
    const float* b1l  = b1  + (size_t)l * 512;
    const float* bnsl = bns + (size_t)l * 512;
    const float* bntl = bnt + (size_t)l * 512;
    const float* w2l  = w2  + (size_t)l * 131072;
    const float* b2l  = b2  + (size_t)l * 256;
    const bool cross = (l & 1);
    const float* src0 = cross ? cur1 : cur0;
    const float* src1 = cross ? cur0 : cur1;

    auto prop = [&](const float* x, const float* src, float* out) {
      gemm_ws<0><<<dim3(4, 8), TPB, 0, stream>>>(qwl, 256, x, x, 256, 1024,
          qbl, nullptr, nullptr, nullptr, Q, 1024, 256, 1024, 256);
      gemm_ws<0><<<dim3(4, 8), TPB, 0, stream>>>(kwl, 256, src, src, 256, 1024,
          kbl, nullptr, nullptr, nullptr, Kb, 1024, 256, 1024, 256);
      gemm_ws<0><<<dim3(4, 8), TPB, 0, stream>>>(vwl, 256, src, src, 256, 1024,
          vbl, nullptr, nullptr, nullptr, Vb, 1024, 256, 1024, 256);
      // scores[h,n,m] = Q_h^T K_h / sqrt(64); head h rows are d = hd*4+h
      gemm_tn<<<dim3(16, 8, 4), TPB, 0, stream>>>(Q, 4096, 1024, Kb, 4096, 1024,
          P, 1024, 1048576, 1024, 1024, 64, 0.125f);
      softmax_rows<<<4096, TPB, 0, stream>>>(P, 1024);
      // msg[hd,n] = sum_m V[hd,m] * prob[n,m]
      gemm_nt<<<dim3(1, 8, 4), TPB, 0, stream>>>(Vb, 4096, 1024, P, 1024, 1048576,
          MSG, 4096, 1024, 64, 1024, 1024);
      gemm_ws<0><<<dim3(4, 8), TPB, 0, stream>>>(mwl, 256, MSG, MSG, 256, 1024,
          mbl, nullptr, nullptr, nullptr, M2, 1024, 256, 1024, 256);
      // h = relu(bn(W1 @ concat(x, msg) + b1)) ; out = x + W2 @ h + b2
      gemm_ws<1><<<dim3(8, 8), TPB, 0, stream>>>(w1l, 512, x, M2, 256, 1024,
          b1l, bnsl, bntl, nullptr, Hb, 1024, 512, 1024, 512);
      gemm_ws<2><<<dim3(4, 8), TPB, 0, stream>>>(w2l, 512, Hb, Hb, 512, 1024,
          b2l, nullptr, nullptr, x, out, 1024, 256, 1024, 512);
    };
    prop(cur0, src0, nxt0);
    prop(cur1, src1, nxt1);
    float* t;
    t = cur0; cur0 = nxt0; nxt0 = t;
    t = cur1; cur1 = nxt1; nxt1 = t;
  }

  // ---- final projection + score matrix (written into couplings interior) ----
  gemm_ws<0><<<dim3(4, 8), TPB, 0, stream>>>(fw, 256, cur0, cur0, 256, 1024,
      fb, nullptr, nullptr, nullptr, md0, 1024, 256, 1024, 256);
  gemm_ws<0><<<dim3(4, 8), TPB, 0, stream>>>(fw, 256, cur1, cur1, 256, 1024,
      fb, nullptr, nullptr, nullptr, md1, 1024, 256, 1024, 256);
  gemm_tn<<<dim3(16, 8, 1), TPB, 0, stream>>>(md0, 1024, 0, md1, 1024, 0,
      cpl, 1025, 0, 1024, 1024, 256, 0.0625f);
  fill_bins<<<4, TPB, 0, stream>>>(cpl, bin);

  // ---- Sinkhorn (log-domain OT), 100 iterations ----
  zero_uv<<<5, TPB, 0, stream>>>(uu, vv);
  for (int it = 0; it < 100; ++it) {
    sink_row<<<1025, TPB, 0, stream>>>(cpl, vv, uu);
    sink_col<<<5, TPB, 0, stream>>>(cpl, uu, vv);
  }

  // ---- mutual matching ----
  row_argmax<<<1024, TPB, 0, stream>>>(cpl, uu, vv, mxr, id0);
  col_argmax<<<1024, TPB, 0, stream>>>(cpl, uu, vv, mxc, id1);
  int*   o0  = (int*)d_out;
  int*   o1  = o0 + 1024;
  float* s0p = (float*)d_out + 2048;
  float* s1p = (float*)d_out + 3072;
  match0<<<4, TPB, 0, stream>>>(mxr, id0, id1, o0, s0p);
  match1<<<4, TPB, 0, stream>>>(id0, id1, o0, s0p, o1, s1p);
}
